// PairStr2Pair_43791486550442
// MI455X (gfx1250) — matrix operand order repaired and checked by reading, not compiled, not run on hardware
//
#include <hip/hip_runtime.h>
#include <cstddef>
#include <cstdint>

// ---------------------------------------------------------------------------
// CDNA5 / gfx1250 implementation of the RF2 pair_str2pair block.
// All GEMM-shaped math uses v_wmma_f32_16x16x32_f16 (wave32) with 2x2
// register blocking per wave (4 wmma per 8 b128 operand loads).
// ---------------------------------------------------------------------------

typedef _Float16 h16;
typedef __attribute__((ext_vector_type(16))) _Float16 v16h;
typedef __attribute__((ext_vector_type(8)))  _Float16 v8h;
typedef __attribute__((ext_vector_type(8)))  float    v8f;

#define LDIM 384
#define M2   (LDIM * LDIM)

union U16 { v16h v; v8h h[2]; };

__device__ __forceinline__ v8f wmma_f16(const v16h& a, const v16h& b, const v8f& c) {
  return __builtin_amdgcn_wmma_f32_16x16x32_f16(false, a, false, b, (short)0, c,
                                                false, false);
}
__device__ __forceinline__ float sigmoidf_(float x) { return 1.f / (1.f + expf(-x)); }

// A-fragment (16x32 f16): element i -> K = (lane<16?0:16)+i
//   => contiguous 16-half run at K0+asel, asel=(lane<16?0:16)
// B-fragment (32x16 f16): lane col = lane&15; element i -> K = (lane<16?0:16)+i
//   => contiguous 16-half run at K0+bsel, bsel=(lane<16?0:16)
// C/D: VGPR v -> row v + (lane<16?0:8), col = lane&15

__device__ __forceinline__ void load_a(U16& a, const h16* p) {
  a.h[0] = *(const v8h*)p;
  a.h[1] = *(const v8h*)(p + 8);
}
__device__ __forceinline__ void load_b(U16& b, const h16* p) {
  b.h[0] = *(const v8h*)p;
  b.h[1] = *(const v8h*)(p + 8);
}

// ---------------------------------------------------------------------------
// Generic dense GEMM:  C[M,N] = act(scale*(A_f16[M,K] @ W_f32[K,N] + bias))
// Block tile 128x64, 8 waves, each wave a 32x32 (2x2) register tile.
// outmode 0: outF32 = r ; 1: outF16 = r ; 2: accum += gate16*r (residual)
// 3: outF32 = r * other32.   Requires M%128==0, N%64==0, K%32==0, K<=256.
// ---------------------------------------------------------------------------
__global__ __launch_bounds__(256) void k_gemm_dense(
    const h16* __restrict__ A, const float* __restrict__ W,
    const float* __restrict__ bias, int M, int N, int K, float scale, int act,
    int outmode, float* __restrict__ outF32, h16* __restrict__ outF16,
    const h16* __restrict__ gate16, const float* __restrict__ other32,
    float* __restrict__ accum) {
  __shared__ h16 Ws[8][4][32][16];  // [kchunk][n-subtile][lane][elem]
  int m0 = blockIdx.y * 128;
  int n0 = blockIdx.x * 64;
  int tid = threadIdx.x;
  int wave = tid >> 5, lane = tid & 31;
  int wm = (wave & 3) * 32;   // 4 m-positions
  int wn = (wave >> 2) * 32;  // 2 n-positions
  // stage W tile (coalesced f32 reads -> fragment-ready f16 LDS layout)
  for (int idx = tid; idx < K * 64; idx += 256) {
    int kk = idx >> 6, nt = idx & 63;
    h16 val = (h16)W[(size_t)kk * N + n0 + nt];
    int kc = kk >> 5, kr = kk & 31;
    Ws[kc][nt >> 4][(nt & 15) + ((kr >> 4) << 4)][kr & 15] = val;
  }
  __syncthreads();
  int asel = (lane < 16) ? 0 : 16;
  const h16* ar0 = A + (size_t)(m0 + wm + (lane & 15)) * K + asel;
  const h16* ar1 = ar0 + (size_t)16 * K;
  int ns = wn >> 4;  // 0 or 2
  v8f acc00 = {}, acc01 = {}, acc10 = {}, acc11 = {};
  int nk = K >> 5;
  for (int kc = 0; kc < nk; ++kc) {
    U16 a0, a1, b0, b1;
    load_a(a0, ar0 + kc * 32);
    load_a(a1, ar1 + kc * 32);
    const h16* b0p = &Ws[kc][ns][lane][0];
    const h16* b1p = &Ws[kc][ns + 1][lane][0];
    b0.h[0] = ((const v8h*)b0p)[0]; b0.h[1] = ((const v8h*)b0p)[1];
    b1.h[0] = ((const v8h*)b1p)[0]; b1.h[1] = ((const v8h*)b1p)[1];
    acc00 = wmma_f16(a0.v, b0.v, acc00);
    acc01 = wmma_f16(a0.v, b1.v, acc01);
    acc10 = wmma_f16(a1.v, b0.v, acc10);
    acc11 = wmma_f16(a1.v, b1.v, acc11);
  }
  int rsel = (lane < 16) ? 0 : 8;
#pragma unroll
  for (int msub = 0; msub < 2; ++msub) {
#pragma unroll
    for (int nsub = 0; nsub < 2; ++nsub) {
      v8f acc = msub ? (nsub ? acc11 : acc10) : (nsub ? acc01 : acc00);
      int n = n0 + wn + nsub * 16 + (lane & 15);
      float bv = bias ? bias[n] : 0.f;
#pragma unroll
      for (int v = 0; v < 8; ++v) {
        int m = m0 + wm + msub * 16 + rsel + v;
        float r = (acc[v] + bv) * scale;
        if (act == 1) r = sigmoidf_(r);
        else if (act == 2) r = fmaxf(r, 0.f);
        size_t idx = (size_t)m * N + n;
        if (outmode == 0) outF32[idx] = r;
        else if (outmode == 1) outF16[idx] = (h16)r;
        else if (outmode == 2) {
          if (gate16) r *= (float)gate16[idx];
          accum[idx] += r;
        } else {
          outF32[idx] = r * other32[idx];
        }
      }
    }
  }
}

// ---------------------------------------------------------------------------
// Triangle einsum on per-channel planar operands (LDS-free), d = blockIdx.z:
//   out[i,j,d] = sum_k lpl[d][i][k] * rpl[d][j][k]
// Block 128x64, wave 32x32.
// ---------------------------------------------------------------------------
__global__ __launch_bounds__(256) void k_tri_einsum(
    const h16* __restrict__ lpl, const h16* __restrict__ rpl,
    float* __restrict__ outRaw) {
  int d = blockIdx.z;
  int m0 = blockIdx.y * 128;
  int n0 = blockIdx.x * 64;
  int tid = threadIdx.x;
  int wave = tid >> 5, lane = tid & 31;
  int wm = (wave & 3) * 32, wn = (wave >> 2) * 32;
  int asel = (lane < 16) ? 0 : 16;
  int bsel = (lane < 16) ? 0 : 16;
  const h16* ar0 = lpl + (size_t)d * M2 + (size_t)(m0 + wm + (lane & 15)) * LDIM + asel;
  const h16* ar1 = ar0 + (size_t)16 * LDIM;
  const h16* br0 = rpl + (size_t)d * M2 + (size_t)(n0 + wn + (lane & 15)) * LDIM + bsel;
  const h16* br1 = br0 + (size_t)16 * LDIM;
  v8f acc00 = {}, acc01 = {}, acc10 = {}, acc11 = {};
#pragma unroll 2
  for (int k0 = 0; k0 < LDIM; k0 += 32) {
    U16 a0, a1, b0, b1;
    load_a(a0, ar0 + k0);
    load_a(a1, ar1 + k0);
    load_b(b0, br0 + k0);
    load_b(b1, br1 + k0);
    acc00 = wmma_f16(a0.v, b0.v, acc00);
    acc01 = wmma_f16(a0.v, b1.v, acc01);
    acc10 = wmma_f16(a1.v, b0.v, acc10);
    acc11 = wmma_f16(a1.v, b1.v, acc11);
  }
  int rsel = (lane < 16) ? 0 : 8;
#pragma unroll
  for (int msub = 0; msub < 2; ++msub) {
#pragma unroll
    for (int nsub = 0; nsub < 2; ++nsub) {
      v8f acc = msub ? (nsub ? acc11 : acc10) : (nsub ? acc01 : acc00);
      int j = n0 + wn + nsub * 16 + (lane & 15);
#pragma unroll
      for (int v = 0; v < 8; ++v) {
        int i = m0 + wm + msub * 16 + rsel + v;
        outRaw[((size_t)i * LDIM + j) * 32 + d] = acc[v];
      }
    }
  }
}

// ---------------------------------------------------------------------------
// Tied attention logits (LDS-free), head = blockIdx.z, planar output:
//   logits[h][i][j] = sum_n sum_k q[pos(n,i),h,k]*kk[pos(n,j),h,k] + bias
// pos(n,i) = trans ? i*L+n : n*L+i.  Block 128x64, wave 32x32.
// ---------------------------------------------------------------------------
__global__ __launch_bounds__(256) void k_attn_logits(
    const h16* __restrict__ q, const h16* __restrict__ kk,
    const float* __restrict__ bproj, float* __restrict__ logits, int trans) {
  int h = blockIdx.z;
  int m0 = blockIdx.y * 128;
  int n0 = blockIdx.x * 64;
  int tid = threadIdx.x;
  int wave = tid >> 5, lane = tid & 31;
  int wm = (wave & 3) * 32, wn = (wave >> 2) * 32;
  int asel = (lane < 16) ? 0 : 16;
  int bsel = (lane < 16) ? 0 : 16;
  int ri0 = m0 + wm + (lane & 15), ri1 = ri0 + 16;
  int cj0 = n0 + wn + (lane & 15), cj1 = cj0 + 16;
  const h16* qb = q + h * 32 + asel;
  const h16* kb = kk + h * 32 + bsel;
  v8f acc00 = {}, acc01 = {}, acc10 = {}, acc11 = {};
  for (int n = 0; n < LDIM; ++n) {
    size_t pa0 = (trans ? (size_t)ri0 * LDIM + n : (size_t)n * LDIM + ri0) * 128;
    size_t pa1 = (trans ? (size_t)ri1 * LDIM + n : (size_t)n * LDIM + ri1) * 128;
    size_t pb0 = (trans ? (size_t)cj0 * LDIM + n : (size_t)n * LDIM + cj0) * 128;
    size_t pb1 = (trans ? (size_t)cj1 * LDIM + n : (size_t)n * LDIM + cj1) * 128;
    U16 a0, a1, b0, b1;
    load_a(a0, qb + pa0);
    load_a(a1, qb + pa1);
    load_b(b0, kb + pb0);
    load_b(b1, kb + pb1);
    acc00 = wmma_f16(a0.v, b0.v, acc00);
    acc01 = wmma_f16(a0.v, b1.v, acc01);
    acc10 = wmma_f16(a1.v, b0.v, acc10);
    acc11 = wmma_f16(a1.v, b1.v, acc11);
  }
  int rsel = (lane < 16) ? 0 : 8;
#pragma unroll
  for (int msub = 0; msub < 2; ++msub) {
#pragma unroll
    for (int nsub = 0; nsub < 2; ++nsub) {
      v8f acc = msub ? (nsub ? acc11 : acc10) : (nsub ? acc01 : acc00);
      int j = (nsub ? cj1 : cj0);
#pragma unroll
      for (int v = 0; v < 8; ++v) {
        int i = m0 + wm + msub * 16 + rsel + v;
        size_t bidx = trans ? ((size_t)j * LDIM + i) : ((size_t)i * LDIM + j);
        logits[(size_t)h * M2 + (size_t)i * LDIM + j] = acc[v] + bproj[bidx * 4 + h];
      }
    }
  }
}

// ---------------------------------------------------------------------------
// Attention output (LDS-free), batched over (h = blockIdx.z, n = blockIdx.y):
//   o[pos(n,i),h,d] = g[pos(n,i),h,d] * sum_j attnw[h][i][j] * vpl[h*32+d][n][j]
// Block 128x32 (4 waves), wave 32x32 (covers all of N=32).
// ---------------------------------------------------------------------------
__global__ __launch_bounds__(128) void k_attn_out(
    const h16* __restrict__ attnw, const h16* __restrict__ vpl,
    const h16* __restrict__ gbuf, h16* __restrict__ obuf, int trans) {
  int h = blockIdx.z;
  int n = blockIdx.y;
  int m0 = blockIdx.x * 128;
  int tid = threadIdx.x;
  int wave = tid >> 5, lane = tid & 31;
  int wm = wave * 32;
  int asel = (lane < 16) ? 0 : 16;
  int bsel = (lane < 16) ? 0 : 16;
  int ri0 = m0 + wm + (lane & 15);
  int d0 = lane & 15, d1 = d0 + 16;
  const h16* ar0 = attnw + (size_t)h * M2 + (size_t)ri0 * LDIM + asel;
  const h16* ar1 = ar0 + (size_t)16 * LDIM;
  const h16* br0 = vpl + ((size_t)(h * 32 + d0) * LDIM + n) * LDIM + bsel;
  const h16* br1 = vpl + ((size_t)(h * 32 + d1) * LDIM + n) * LDIM + bsel;
  v8f acc00 = {}, acc01 = {}, acc10 = {}, acc11 = {};
#pragma unroll 2
  for (int k0 = 0; k0 < LDIM; k0 += 32) {
    U16 a0, a1, b0, b1;
    load_a(a0, ar0 + k0);
    load_a(a1, ar1 + k0);
    load_b(b0, br0 + k0);
    load_b(b1, br1 + k0);
    acc00 = wmma_f16(a0.v, b0.v, acc00);
    acc01 = wmma_f16(a0.v, b1.v, acc01);
    acc10 = wmma_f16(a1.v, b0.v, acc10);
    acc11 = wmma_f16(a1.v, b1.v, acc11);
  }
  int rsel = (lane < 16) ? 0 : 8;
#pragma unroll
  for (int msub = 0; msub < 2; ++msub) {
#pragma unroll
    for (int nsub = 0; nsub < 2; ++nsub) {
      v8f acc = msub ? (nsub ? acc11 : acc10) : (nsub ? acc01 : acc00);
      int dcol = (nsub ? d1 : d0);
#pragma unroll
      for (int v = 0; v < 8; ++v) {
        int i = m0 + wm + msub * 16 + rsel + v;
        size_t pos = trans ? ((size_t)i * LDIM + n) : ((size_t)n * LDIM + i);
        size_t idx = pos * 128 + h * 32 + dcol;
        obuf[idx] = (h16)(acc[v] * (float)gbuf[idx]);
      }
    }
  }
}

// ---------------------------------------------------------------------------
// Elementwise / reduction kernels
// ---------------------------------------------------------------------------
__global__ __launch_bounds__(256) void k_ln_f16(const float* __restrict__ x,
                                                h16* __restrict__ y,
                                                const float* __restrict__ g,
                                                const float* __restrict__ b,
                                                int C) {
  int wid = blockIdx.x * 8 + (threadIdx.x >> 5);
  int lane = threadIdx.x & 31;
  const float* xr = x + (size_t)wid * C;
  int per = C >> 5;  // C in {32,128}
  float vals[4];
  float s = 0.f;
  for (int i = 0; i < per; ++i) { vals[i] = xr[lane + (i << 5)]; s += vals[i]; }
  for (int m = 16; m > 0; m >>= 1) s += __shfl_xor(s, m, 32);
  float mu = s / (float)C;
  float v2 = 0.f;
  for (int i = 0; i < per; ++i) { float d = vals[i] - mu; v2 += d * d; }
  for (int m = 16; m > 0; m >>= 1) v2 += __shfl_xor(v2, m, 32);
  float rstd = rsqrtf(v2 / (float)C + 1e-5f);
  h16* yr = y + (size_t)wid * C;
  for (int i = 0; i < per; ++i) {
    int c = lane + (i << 5);
    yr[c] = (h16)((vals[i] - mu) * rstd * g[c] + b[c]);
  }
}

__global__ __launch_bounds__(256) void k_state_proj(
    const float* __restrict__ st, const float* g, const float* b,
    const float* Wl, const float* bl, const float* Wr, const float* br,
    float* leftS, float* rightS) {
  __shared__ float sx[8][32];
  int w = threadIdx.x >> 5, lane = threadIdx.x & 31;
  int row = blockIdx.x * 8 + w;  // grid = 48 exact
  float v = st[row * 32 + lane];
  float s = v;
  for (int m = 16; m > 0; m >>= 1) s += __shfl_xor(s, m, 32);
  float mu = s * (1.f / 32.f);
  float d = v - mu;
  float q = d * d;
  for (int m = 16; m > 0; m >>= 1) q += __shfl_xor(q, m, 32);
  float rstd = rsqrtf(q * (1.f / 32.f) + 1e-5f);
  sx[w][lane] = d * rstd * g[lane] + b[lane];
  __syncthreads();
  int j = lane & 15;
  const float* W = (lane < 16) ? Wl : Wr;
  const float* bb = (lane < 16) ? bl : br;
  float acc = bb[j];
  for (int c = 0; c < 32; ++c) acc += sx[w][c] * W[c * 16 + j];
  if (lane < 16) leftS[row * 16 + j] = acc;
  else           rightS[row * 16 + j] = acc;
}

__global__ __launch_bounds__(256) void k_outer(const float* __restrict__ leftS,
                                               const float* __restrict__ rightS,
                                               h16* __restrict__ O) {
  size_t gid = (size_t)blockIdx.x * 256 + threadIdx.x;  // grid = M2 exact
  int c = (int)(gid & 255);
  size_t pm = gid >> 8;
  int l = (int)(pm / LDIM), m = (int)(pm % LDIM);
  O[gid] = (h16)(leftS[l * 16 + (c >> 4)] * rightS[m * 16 + (c & 15)]);
}

__global__ __launch_bounds__(256) void k_f16copy(const float* __restrict__ x,
                                                 h16* __restrict__ y) {
  size_t i = (size_t)blockIdx.x * 256 + threadIdx.x;
  y[i] = (h16)x[i];
}

// pack 4x (128x32) weights / (32) biases into one 128x128 / 128
__global__ __launch_bounds__(256) void k_packW4(const float* __restrict__ W0,
                                                const float* __restrict__ W1,
                                                const float* __restrict__ W2,
                                                const float* __restrict__ W3,
                                                float* __restrict__ Wcat) {
  int idx = blockIdx.x * 256 + threadIdx.x;  // 128*128 exact
  int k = idx >> 7, c = idx & 127;
  int qsel = c >> 5, n = c & 31;
  const float* Wq = qsel == 0 ? W0 : qsel == 1 ? W1 : qsel == 2 ? W2 : W3;
  Wcat[idx] = Wq[k * 32 + n];
}
__global__ __launch_bounds__(128) void k_packb4(const float* __restrict__ b0,
                                                const float* __restrict__ b1,
                                                const float* __restrict__ b2,
                                                const float* __restrict__ b3,
                                                float* __restrict__ bcat) {
  int c = threadIdx.x;
  int qsel = c >> 5, n = c & 31;
  const float* bq = qsel == 0 ? b0 : qsel == 1 ? b1 : qsel == 2 ? b2 : b3;
  bcat[c] = bq[n];
}

// combine packed tri projections -> planar operands (transpose folded)
// rawcat[pos][0:32]=lgate, [32:64]=left, [64:96]=rgate, [96:128]=right
__global__ __launch_bounds__(256) void k_tri_combine(
    const float* __restrict__ rawcat, h16* __restrict__ lpl,
    h16* __restrict__ rpl, int incoming) {
  size_t id = (size_t)blockIdx.x * 256 + threadIdx.x;  // M2*32 exact
  int d = (int)(id & 31);
  size_t pos = id >> 5;
  const float* rowp = rawcat + pos * 128;
  size_t dst;
  if (incoming) {
    size_t a = pos / LDIM, bq = pos % LDIM;  // pos = k*L + i -> [i][k]
    dst = (size_t)d * M2 + bq * LDIM + a;
  } else {
    dst = (size_t)d * M2 + pos;              // pos = i*L + k
  }
  lpl[dst] = (h16)(sigmoidf_(rowp[d]) * rowp[32 + d]);
  rpl[dst] = (h16)(sigmoidf_(rowp[64 + d]) * rowp[96 + d] * (1.f / (float)LDIM));
}

// v re-layout: vpl[(h*32+d)][n][j] = v[pos(n,j)*128 + h*32 + d]
__global__ __launch_bounds__(256) void k_vplanar(const h16* __restrict__ vbuf,
                                                 h16* __restrict__ vpl, int trans) {
  size_t id = (size_t)blockIdx.x * 256 + threadIdx.x;  // M2*128 exact
  int j = (int)(id % LDIM);
  size_t r = id / LDIM;
  int n = (int)(r % LDIM);
  int hd = (int)(r / LDIM);  // h*32+d
  size_t pos = trans ? (size_t)j * LDIM + n : (size_t)n * LDIM + j;
  vpl[id] = vbuf[pos * 128 + hd];
}

__global__ __launch_bounds__(256) void k_bias_proj(const h16* __restrict__ bln,
                                                   const float* __restrict__ Wb,
                                                   float* __restrict__ bproj) {
  int p = blockIdx.x * 8 + (threadIdx.x >> 5);  // grid = M2/8 exact
  int lane = threadIdx.x & 31;
  float acc[4] = {0.f, 0.f, 0.f, 0.f};
  for (int t = 0; t < 4; ++t) {
    int c = lane + t * 32;
    float x = (float)bln[(size_t)p * 128 + c];
    for (int h = 0; h < 4; ++h) acc[h] += x * Wb[c * 4 + h];
  }
  for (int h = 0; h < 4; ++h)
    for (int m = 16; m > 0; m >>= 1) acc[h] += __shfl_xor(acc[h], m, 32);
  if (lane < 4) bproj[(size_t)p * 4 + lane] = acc[lane];
}

// softmax over j; planar [h][i][j] in and out
__global__ __launch_bounds__(256) void k_softmax(const float* __restrict__ logits,
                                                 h16* __restrict__ attnw) {
  int row = blockIdx.x * 8 + (threadIdx.x >> 5);  // (h,i): grid = 1536/8 exact
  int h = row / LDIM, i = row % LDIM;
  int lane = threadIdx.x & 31;
  const float* lr = logits + (size_t)h * M2 + (size_t)i * LDIM;
  h16* ar = attnw + (size_t)h * M2 + (size_t)i * LDIM;
  float v[12];
  float mx = -1e30f;
  for (int t = 0; t < 12; ++t) {
    v[t] = lr[lane + t * 32];
    mx = fmaxf(mx, v[t]);
  }
  for (int m = 16; m > 0; m >>= 1) mx = fmaxf(mx, __shfl_xor(mx, m, 32));
  float s = 0.f;
  for (int t = 0; t < 12; ++t) { v[t] = expf(v[t] - mx); s += v[t]; }
  for (int m = 16; m > 0; m >>= 1) s += __shfl_xor(s, m, 32);
  float inv = 1.f / s;
  for (int t = 0; t < 12; ++t) ar[lane + t * 32] = (h16)(v[t] * inv);
}

// ---------------------------------------------------------------------------
// Host side
// ---------------------------------------------------------------------------
static inline void launch_gemm(hipStream_t s, const h16* A, const float* W,
                               const float* bias, int M, int N, int K, float scale,
                               int act, int outmode, float* o32, h16* o16,
                               const h16* gate16, const float* other32, float* accum) {
  dim3 grid(N / 64, M / 128, 1);
  k_gemm_dense<<<grid, 256, 0, s>>>(A, W, bias, M, N, K, scale, act, outmode, o32,
                                    o16, gate16, other32, accum);
}

extern "C" void kernel_launch(void* const* d_in, const int* in_sizes, int n_in,
                              void* d_out, int out_size, void* d_ws, size_t ws_size,
                              hipStream_t stream) {
  (void)in_sizes; (void)n_in; (void)out_size; (void)ws_size;
  const float* pair_in = (const float*)d_in[0];
  const float* rbf_in  = (const float*)d_in[1];
  const float* state   = (const float*)d_in[2];

  // jax pytree leaves (sorted dict keys), starting at index 3:
  //  col_attn (12): 3 b.W, 4 g.W, 5 g.b, 6 k.W, 7 nb.b, 8 nb.g, 9 np.b,
  //                 10 np.g, 11 out.W, 12 out.b, 13 q.W, 14 v.W
  //  emb_rbf: 15 W, 16 b
  //  ff: 17 l1.W, 18 l1.b, 19 l2.W, 20 l2.b, 21 n.b, 22 n.g
  //  norm_state: 23 b, 24 g ; proj_left: 25 W, 26 b ; proj_right: 27 W, 28 b
  //  row_attn (12): 29..40 ; to_gate: 41 W, 42 b
  //  tri_in (16): 43..58 ; tri_out (16): 59..74
  //    (+0 gate.W +1 gate.b +2 left.W +3 left.b +4 lgate.W +5 lgate.b
  //     +6 norm.b +7 norm.g +8 nout.b +9 nout.g +10 out.W +11 out.b
  //     +12 rgate.W +13 rgate.b +14 right.W +15 right.b)
  //  crop: 75
  auto F = [&](int i) { return (const float*)d_in[i]; };

  float* pair = (float*)d_out;  // running pair accumulator

  // ---- workspace carve ---------------------------------------------------
  char* ws = (char*)d_ws;
  size_t off = 0;
  auto alloc = [&](size_t bytes) -> void* {
    void* p = ws + off;
    off += (bytes + 255) & ~(size_t)255;
    return p;
  };
  h16*   xln    = (h16*)alloc((size_t)M2 * 128 * 2);
  h16*   bln    = (h16*)alloc((size_t)M2 * 128 * 2);
  h16*   gpair  = (h16*)alloc((size_t)M2 * 128 * 2);
  h16*   qbuf   = (h16*)alloc((size_t)M2 * 128 * 2);
  h16*   kbuf   = (h16*)alloc((size_t)M2 * 128 * 2);
  h16*   vbuf   = (h16*)alloc((size_t)M2 * 128 * 2);
  h16*   gbuf   = (h16*)alloc((size_t)M2 * 128 * 2);
  h16*   obuf   = (h16*)alloc((size_t)M2 * 128 * 2);
  float* rbfe   = (float*)alloc((size_t)M2 * 128 * 4);  // reused: packed tri raw
  float* rbfg   = (float*)alloc((size_t)M2 * 128 * 4);
  h16*   Obig   = (h16*)alloc((size_t)M2 * 256 * 2);    // reused: vpl / ff hidden
  float* triraw = (float*)alloc((size_t)M2 * 32 * 4);
  h16*   lpl    = (h16*)alloc((size_t)M2 * 32 * 2);     // planar [d][i][k]
  h16*   rpl    = (h16*)alloc((size_t)M2 * 32 * 2);     // planar [d][j][k]
  h16*   triln  = (h16*)alloc((size_t)M2 * 32 * 2);
  float* logits = (float*)alloc((size_t)M2 * 4 * 4);    // planar [h][i][j]
  h16*   attnw  = (h16*)alloc((size_t)M2 * 4 * 2);      // planar [h][i][j]
  float* bproj  = (float*)alloc((size_t)M2 * 4 * 4);
  float* leftS  = (float*)alloc((size_t)LDIM * 16 * 4);
  float* rightS = (float*)alloc((size_t)LDIM * 16 * 4);
  h16*   rbf16  = (h16*)alloc((size_t)M2 * 64 * 2);
  float* Wcat   = (float*)alloc((size_t)128 * 128 * 4);
  float* bcat   = (float*)alloc((size_t)128 * 4);
  float* rawcat = rbfe;  // M2 x 128 f32 (free after prelude)
  h16* vpl = Obig;       // [h*32+d][n][j] (Obig free after prelude)
  h16* ffh = Obig;       // M2 x 256 f16 (ff runs last)

  // ---- seed running pair -------------------------------------------------
  hipMemcpyAsync(pair, pair_in, (size_t)M2 * 128 * 4, hipMemcpyDeviceToDevice, stream);

  // ---- prelude: rbfg = sigmoid(outer(l,r)@Wg+bg) * (rbf@We+be) ----------
  k_state_proj<<<LDIM / 8, 256, 0, stream>>>(state, F(24), F(23), F(25), F(26),
                                             F(27), F(28), leftS, rightS);
  k_outer<<<M2, 256, 0, stream>>>(leftS, rightS, Obig);
  k_f16copy<<<(int)((size_t)M2 * 64 / 256), 256, 0, stream>>>(rbf_in, rbf16);
  launch_gemm(stream, rbf16, F(15), F(16), M2, 128, 64, 1.f, 0, 0, rbfe, nullptr,
              nullptr, nullptr, nullptr);
  launch_gemm(stream, Obig, F(41), F(42), M2, 128, 256, 1.f, 1, 3, rbfg, nullptr,
              nullptr, rbfe, nullptr);

  // ---- triangle multiplication stages ------------------------------------
  auto tri_stage = [&](int pb, int incoming) {
    k_ln_f16<<<M2 / 8, 256, 0, stream>>>(pair, xln, F(pb + 7), F(pb + 6), 128);
    // pack {lgate,left,rgate,right} into one 128->128 GEMM
    k_packW4<<<64, 256, 0, stream>>>(F(pb + 4), F(pb + 2), F(pb + 12), F(pb + 14),
                                     Wcat);
    k_packb4<<<1, 128, 0, stream>>>(F(pb + 5), F(pb + 3), F(pb + 13), F(pb + 15),
                                    bcat);
    launch_gemm(stream, xln, Wcat, bcat, M2, 128, 128, 1.f, 0, 0, rawcat, nullptr,
                nullptr, nullptr, nullptr);
    k_tri_combine<<<(int)((size_t)M2 * 32 / 256), 256, 0, stream>>>(rawcat, lpl,
                                                                    rpl, incoming);
    launch_gemm(stream, xln, F(pb + 0), F(pb + 1), M2, 128, 128, 1.f, 1, 1,
                nullptr, gpair, nullptr, nullptr, nullptr);  // sigmoid(gate)
    k_tri_einsum<<<dim3(LDIM / 64, LDIM / 128, 32), 256, 0, stream>>>(lpl, rpl,
                                                                      triraw);
    k_ln_f16<<<M2 / 8, 256, 0, stream>>>(triraw, triln, F(pb + 9), F(pb + 8), 32);
    launch_gemm(stream, triln, F(pb + 10), F(pb + 11), M2, 128, 32, 1.f, 0, 2,
                nullptr, nullptr, gpair, nullptr, pair);  // gate*out, residual
  };
  tri_stage(59, 0);  // tri_out (outgoing)
  tri_stage(43, 1);  // tri_in  (incoming)

  // ---- axial attention stages --------------------------------------------
  auto attn_stage = [&](int pb, int trans) {
    // +0 b.W +1 g.W +2 g.b +3 k.W +4 nb.b +5 nb.g +6 np.b +7 np.g
    // +8 out.W +9 out.b +10 q.W +11 v.W
    k_ln_f16<<<M2 / 8, 256, 0, stream>>>(pair, xln, F(pb + 7), F(pb + 6), 128);
    k_ln_f16<<<M2 / 8, 256, 0, stream>>>(rbfg, bln, F(pb + 5), F(pb + 4), 128);
    k_bias_proj<<<M2 / 8, 256, 0, stream>>>(bln, F(pb + 0), bproj);
    launch_gemm(stream, xln, F(pb + 10), nullptr, M2, 128, 128, 0.1767766953f, 0,
                1, nullptr, qbuf, nullptr, nullptr, nullptr);  // q / sqrt(32)
    launch_gemm(stream, xln, F(pb + 3), nullptr, M2, 128, 128, 1.f / (float)LDIM,
                0, 1, nullptr, kbuf, nullptr, nullptr, nullptr);  // k / L
    launch_gemm(stream, xln, F(pb + 11), nullptr, M2, 128, 128, 1.f, 0, 1,
                nullptr, vbuf, nullptr, nullptr, nullptr);  // v
    launch_gemm(stream, xln, F(pb + 1), F(pb + 2), M2, 128, 128, 1.f, 1, 1,
                nullptr, gbuf, nullptr, nullptr, nullptr);  // sigmoid(g)
    k_vplanar<<<(int)((size_t)M2 * 128 / 256), 256, 0, stream>>>(vbuf, vpl, trans);
    k_attn_logits<<<dim3(LDIM / 64, LDIM / 128, 4), 256, 0, stream>>>(
        qbuf, kbuf, bproj, logits, trans);
    k_softmax<<<(LDIM * 4) / 8, 256, 0, stream>>>(logits, attnw);
    k_attn_out<<<dim3(LDIM / 128, LDIM, 4), 128, 0, stream>>>(attnw, vpl, gbuf,
                                                              obuf, trans);
    launch_gemm(stream, obuf, F(pb + 8), F(pb + 9), M2, 128, 128, 1.f, 0, 2,
                nullptr, nullptr, nullptr, nullptr, pair);  // residual add
  };
  attn_stage(29, 1);  // row attention (transposed view)
  attn_stage(3, 0);   // col attention

  // ---- feed forward -------------------------------------------------------
  k_ln_f16<<<M2 / 8, 256, 0, stream>>>(pair, xln, F(22), F(21), 128);
  launch_gemm(stream, xln, F(17), F(18), M2, 256, 128, 1.f, 2, 1, nullptr, ffh,
              nullptr, nullptr, nullptr);  // relu(lin1)
  launch_gemm(stream, ffh, F(19), F(20), M2, 128, 256, 1.f, 0, 2, nullptr,
              nullptr, nullptr, nullptr, pair);  // lin2, residual add
}
